// NNMFConv2dP_19902878450290
// MI455X (gfx1250) — matrix operand order repaired and checked by reading, not compiled, not run on hardware
//
#include <hip/hip_runtime.h>

typedef __attribute__((ext_vector_type(16))) __bf16 v16bf;
typedef __attribute__((ext_vector_type(8)))  __bf16 v8bf;
typedef __attribute__((ext_vector_type(8)))  float  v8f;

#define CIN   256
#define COUT  512
#define NPIX  32768   // 32 * 32 * 32
#define NITER 20

// Per-wave LDS slice layout (bytes):
//   h  : 16*512 f32  = 32768   (unnormalized master copy)
//   x  : 16*256 f32  = 16384
//   q  : 16*256 bf16 =  8192   (C-layout -> A-layout transpose staging)
//   rs : 16 f32      =    64   (1/rowsum, indexed by pixel row)
static constexpr int OFF_H  = 0;
static constexpr int OFF_X  = 32768;
static constexpr int OFF_Q  = 32768 + 16384;
static constexpr int OFF_RS = 32768 + 16384 + 8192;
static constexpr int WAVE_LDS_BYTES = 57472;   // padded; 4 waves -> 229888 B < 320 KB

// ---------------------------------------------------------------------------
// Prep: pack weight[COUT][CIN] (f32) into bf16 B-operand layouts.
//  W1 (for rec = h @ W):    B1[k][n] = weight[k][n],  k over COUT, n over CIN
//  W2 (for t = q @ W^T):    B2[k][n] = weight[n][k],  k over CIN,  n over COUT
// B-operand lane layout (16x16x32 bf16): lane holds 16 K for column n = lane&15;
// lanes 0-15 -> K base+0..7,16..23, lanes 16-31 -> K base+8..15,24..31.
// ---------------------------------------------------------------------------
__global__ void pack_weights_kernel(const float* __restrict__ w,
                                    __bf16* __restrict__ w1,
                                    __bf16* __restrict__ w2) {
    int tid = blockIdx.x * blockDim.x + threadIdx.x;   // 0..16383
    if (tid < 8192) {
        // W1: vector index = (kc*16 + nt)*32 + lane,  kc<16, nt<16
        int lane = tid & 31;
        int t    = tid >> 5;
        int nt   = t & 15;
        int kc   = t >> 4;
        int n    = nt * 16 + (lane & 15);
        int kb   = kc * 32 + ((lane >> 4) << 3);
        v16bf o;
#pragma unroll
        for (int i = 0; i < 16; ++i) o[i] = (__bf16)w[(kb + i + (i >= 8 ? 8 : 0)) * CIN + n];
        *(v16bf*)(w1 + (size_t)tid * 16) = o;
    } else if (tid < 16384) {
        // W2: vector index = (kc2*32 + nt2)*32 + lane,  kc2<8, nt2<32
        int id   = tid - 8192;
        int lane = id & 31;
        int t    = id >> 5;
        int nt2  = t & 31;
        int kc2  = t >> 5;
        int n    = nt2 * 16 + (lane & 15);
        int kb   = kc2 * 32 + ((lane >> 4) << 3);
        v16bf o;
#pragma unroll
        for (int i = 0; i < 16; ++i) o[i] = (__bf16)w[n * CIN + kb + i + (i >= 8 ? 8 : 0)];
        *(v16bf*)(w2 + (size_t)id * 16) = o;
    }
}

// Build a bf16 A-operand vector from 16 pre-scaled f32 values.
static __device__ inline v16bf bf16x16_from_f32(float4 a, float4 b, float4 c, float4 d) {
    v16bf r;
    r[0]  = (__bf16)a.x; r[1]  = (__bf16)a.y; r[2]  = (__bf16)a.z; r[3]  = (__bf16)a.w;
    r[4]  = (__bf16)b.x; r[5]  = (__bf16)b.y; r[6]  = (__bf16)b.z; r[7]  = (__bf16)b.w;
    r[8]  = (__bf16)c.x; r[9]  = (__bf16)c.y; r[10] = (__bf16)c.z; r[11] = (__bf16)c.w;
    r[12] = (__bf16)d.x; r[13] = (__bf16)d.y; r[14] = (__bf16)d.z; r[15] = (__bf16)d.w;
    return r;
}

static __device__ inline float4 scale4(float4 a, float s) {
    a.x *= s; a.y *= s; a.z *= s; a.w *= s;
    return a;
}

// ---------------------------------------------------------------------------
// Main kernel: one wave32 per 16-pixel block; h/x/q resident in LDS for all
// 20 iterations; weights streamed from L2 in packed bf16 B-operand form.
// A operands are built once per iteration into registers (Ah[16], Aq[8]);
// row-normalization is folded into the A conversion and the h update.
// ---------------------------------------------------------------------------
__global__ __launch_bounds__(128, 1)
void nnmf_resident_kernel(const float* __restrict__ input,
                          const __bf16* __restrict__ w1,
                          const __bf16* __restrict__ w2,
                          float* __restrict__ out) {
    extern __shared__ char smem[];
    const int lane = threadIdx.x & 31;
    const int wid  = threadIdx.x >> 5;
    const int blk  = blockIdx.x * 4 + wid;        // 0..2047 (NPIX/16 blocks)
    const int bimg = blk >> 6;                    // batch index (64 blocks per image)
    const int pbase = (blk & 63) * 16;            // pixel offset within image plane

    char* wb = smem + wid * WAVE_LDS_BYTES;
    float*  hL  = (float*)(wb + OFF_H);           // [16][512] f32, unnormalized
    float*  xL  = (float*)(wb + OFF_X);           // [16][256] f32
    __bf16* qL  = (__bf16*)(wb + OFF_Q);          // [16][256] bf16
    float*  rsL = (float*)(wb + OFF_RS);          // [16] f32 (1/rowsum)

    const int hi = lane >> 4;                     // half-wave id (0/1)
    const int ln = lane & 15;

    // ---- load x tile [16 pixels][256 channels] (one-time HBM read) ----
    {
        const float* src = input + (size_t)bimg * CIN * 1024 + pbase;
        for (int c0 = 0; c0 < CIN; c0 += 2) {
            int c = c0 + hi;
            xL[ln * CIN + c] = src[(size_t)c * 1024 + ln];
        }
    }
    // ---- init: hL = 1.0 (unnormalized), rowsum = COUT -> inv = 1/COUT ----
    {
        float4 v; v.x = v.y = v.z = v.w = 1.0f;
        float4* hv = (float4*)hL;
        for (int i = lane; i < (16 * COUT) / 4; i += 32) hv[i] = v;
        if (lane < 16) rsL[lane] = 1.0f / (float)COUT;
    }
    float invs[8];                                // 1/rowsum for rows hi*8+v
#pragma unroll
    for (int v = 0; v < 8; ++v) invs[v] = 1.0f / (float)COUT;
    __syncthreads();

    const float* hrow = hL + ln * COUT;
    const __bf16* qrow = qL + ln * CIN;

    for (int it = 0; it < NITER; ++it) {
        // ==== build normalized bf16 A operands (GEMM1) in registers ========
        const float invA = rsL[ln];
        v16bf Ah[16];
#pragma unroll
        for (int kc = 0; kc < 16; ++kc) {
            const int kb = kc * 32 + (hi ? 8 : 0);
            float4 f0 = scale4(*(const float4*)(hrow + kb),      invA);
            float4 f1 = scale4(*(const float4*)(hrow + kb + 4),  invA);
            float4 f2 = scale4(*(const float4*)(hrow + kb + 16), invA);
            float4 f3 = scale4(*(const float4*)(hrow + kb + 20), invA);
            Ah[kc] = bf16x16_from_f32(f0, f1, f2, f3);
        }

        // ==== GEMM1: rec = h_norm @ W ; q = x/(rec+eps) ====================
        for (int ntp = 0; ntp < 8; ++ntp) {
            v8f acc0 = {};
            v8f acc1 = {};
#pragma unroll
            for (int kc = 0; kc < 16; ++kc) {
                v16bf B0 = *(const v16bf*)(w1 + (size_t)((kc * 16 + ntp * 2    ) * 32 + lane) * 16);
                v16bf B1 = *(const v16bf*)(w1 + (size_t)((kc * 16 + ntp * 2 + 1) * 32 + lane) * 16);
                acc0 = __builtin_amdgcn_wmma_f32_16x16x32_bf16(false, Ah[kc], false, B0,
                                                               (short)0, acc0, false, false);
                acc1 = __builtin_amdgcn_wmma_f32_16x16x32_bf16(false, Ah[kc], false, B1,
                                                               (short)0, acc1, false, false);
            }
#pragma unroll
            for (int v = 0; v < 8; ++v) {
                int m  = hi * 8 + v;
                int n0 = (ntp * 2) * 16 + ln;
                int n1 = n0 + 16;
                float r0 = __builtin_amdgcn_rcpf(acc0[v] + 1e-20f);
                float r1 = __builtin_amdgcn_rcpf(acc1[v] + 1e-20f);
                qL[m * CIN + n0] = (__bf16)(xL[m * CIN + n0] * r0);
                qL[m * CIN + n1] = (__bf16)(xL[m * CIN + n1] * r1);
            }
        }

        // ==== load q A operands (GEMM2) into registers =====================
        v16bf Aq[8];
#pragma unroll
        for (int kc = 0; kc < 8; ++kc) {
            const int kb = kc * 32 + (hi ? 8 : 0);
            v8bf lo  = *(const v8bf*)(qrow + kb);
            v8bf hi8 = *(const v8bf*)(qrow + kb + 16);
            Aq[kc] = __builtin_shufflevector(lo, hi8,
                         0, 1, 2, 3, 4, 5, 6, 7, 8, 9, 10, 11, 12, 13, 14, 15);
        }

        // ==== GEMM2: t = q @ W^T ; h_new = h_norm * t ; row sums ===========
        float psum[8];
#pragma unroll
        for (int v = 0; v < 8; ++v) psum[v] = 0.0f;

        for (int ntp = 0; ntp < 16; ++ntp) {
            v8f acc0 = {};
            v8f acc1 = {};
#pragma unroll
            for (int kc = 0; kc < 8; ++kc) {
                v16bf B0 = *(const v16bf*)(w2 + (size_t)((kc * 32 + ntp * 2    ) * 32 + lane) * 16);
                v16bf B1 = *(const v16bf*)(w2 + (size_t)((kc * 32 + ntp * 2 + 1) * 32 + lane) * 16);
                acc0 = __builtin_amdgcn_wmma_f32_16x16x32_bf16(false, Aq[kc], false, B0,
                                                               (short)0, acc0, false, false);
                acc1 = __builtin_amdgcn_wmma_f32_16x16x32_bf16(false, Aq[kc], false, B1,
                                                               (short)0, acc1, false, false);
            }
#pragma unroll
            for (int v = 0; v < 8; ++v) {
                int m  = hi * 8 + v;
                int n0 = (ntp * 2) * 16 + ln;
                int n1 = n0 + 16;
                float h0 = hL[m * COUT + n0] * invs[v] * acc0[v];
                float h1 = hL[m * COUT + n1] * invs[v] * acc1[v];
                hL[m * COUT + n0] = h0;
                hL[m * COUT + n1] = h1;
                psum[v] += h0 + h1;
            }
        }

        // ==== row-sum reduce across the 16 lanes of each half-wave =========
#pragma unroll
        for (int v = 0; v < 8; ++v) {
            float s = psum[v];
            s += __shfl_xor(s, 1, 32);
            s += __shfl_xor(s, 2, 32);
            s += __shfl_xor(s, 4, 32);
            s += __shfl_xor(s, 8, 32);
            float inv = __builtin_amdgcn_rcpf(s + 1e-19f);
            invs[v] = inv;                         // update-phase mapping (rows hi*8+v)
            if (ln == 0) rsL[hi * 8 + v] = inv;    // A-build mapping (row = ln)
        }
        __syncthreads();
    }

    // ---- store out [B][COUT][H*W], applying final normalization ----
    {
        const float inv = rsL[ln];
        float* dst = out + (size_t)bimg * COUT * 1024 + pbase;
        for (int c0 = 0; c0 < COUT; c0 += 2) {
            int c = c0 + hi;
            dst[(size_t)c * 1024 + ln] = hL[ln * COUT + c] * inv;
        }
    }
}

extern "C" void kernel_launch(void* const* d_in, const int* in_sizes, int n_in,
                              void* d_out, int out_size, void* d_ws, size_t ws_size,
                              hipStream_t stream) {
    const float* input  = (const float*)d_in[0];   // [32,256,32,32] f32
    const float* weight = (const float*)d_in[1];   // [512,256] f32
    float* out = (float*)d_out;                    // [32,512,32,32] f32

    __bf16* w1 = (__bf16*)d_ws;                           // 256 KB packed B for GEMM1
    __bf16* w2 = (__bf16*)((char*)d_ws + 256 * 1024);     // 256 KB packed B for GEMM2

    pack_weights_kernel<<<64, 256, 0, stream>>>(weight, w1, w2);

    const size_t lds = 4 * (size_t)WAVE_LDS_BYTES;        // ~225 KB per workgroup
    (void)hipFuncSetAttribute((const void*)nnmf_resident_kernel,
                              hipFuncAttributeMaxDynamicSharedMemorySize, (int)lds);
    nnmf_resident_kernel<<<NPIX / 16 / 4, 128, lds, stream>>>(input, w1, w2, out);
}
